// Block_63282048139761
// MI455X (gfx1250) — matrix operand order repaired and checked by reading, not compiled, not run on hardware
//
#include <hip/hip_runtime.h>
#include <math.h>

// ---------------- types ----------------
typedef __attribute__((ext_vector_type(16))) __bf16 v16bf;
typedef __attribute__((ext_vector_type(8)))  __bf16 v8bf;
typedef __attribute__((ext_vector_type(8)))  float  v8f;

static constexpr int  TB   = 4;          // batch
static constexpr int  TT   = 768;        // time
static constexpr int  CC   = 2048;       // channels
static constexpr int  HH   = 32;         // heads
static constexpr int  NH   = 64;         // head size
static constexpr int  MM   = TB * TT;    // 3072 rows
static constexpr int  C4   = 4 * CC;     // 8192

__device__ __forceinline__ __bf16 f2bf(float f) {
  union { float f; unsigned u; } x; x.f = f;
  unsigned r = x.u + 0x7FFFu + ((x.u >> 16) & 1u);
  unsigned short h = (unsigned short)(r >> 16);
  __bf16 b; __builtin_memcpy(&b, &h, 2); return b;
}
__device__ __forceinline__ float sigmf(float x) { return 1.0f / (1.0f + __expf(-x)); }

// ---------------- elementwise kernels ----------------
__global__ void k_f32_to_bf16(const float* __restrict__ s, __bf16* __restrict__ d, int n) {
  int i = blockIdx.x * blockDim.x + threadIdx.x;
  if (i < n) d[i] = f2bf(s[i]);
}

__global__ void k_tanh_bf(const float* __restrict__ s, __bf16* __restrict__ d, int n) {
  int i = blockIdx.x * blockDim.x + threadIdx.x;
  if (i < n) d[i] = f2bf(tanhf(s[i]));
}

__global__ void k_sigmoid_bf(const float* __restrict__ s, __bf16* __restrict__ d, int n) {
  int i = blockIdx.x * blockDim.x + threadIdx.x;
  if (i < n) d[i] = f2bf(sigmf(s[i]));
}

__global__ void k_relusq_bf(const float* __restrict__ s, __bf16* __restrict__ d, int n) {
  int i = blockIdx.x * blockDim.x + threadIdx.x;
  if (i < n) { float t = s[i]; t = t > 0.f ? t * t : 0.f; d[i] = f2bf(t); }
}

__global__ void k_wlog(const float* __restrict__ t, const float* __restrict__ w0,
                       float* __restrict__ wl, int n) {
  int i = blockIdx.x * blockDim.x + threadIdx.x;
  if (i < n) {
    int c = i % CC;
    float z = -(w0[c] + t[i]);
    float sp = (z > 20.f) ? z : log1pf(__expf(z));
    wl[i] = -sp - 0.5f;
  }
}

__global__ void k_a_and_k(const float* __restrict__ t, const float* __restrict__ a0,
                          const float* __restrict__ k_a, float* __restrict__ a,
                          float* __restrict__ k, int n) {
  int i = blockIdx.x * blockDim.x + threadIdx.x;
  if (i < n) {
    int c = i % CC;
    float av = sigmf(a0[c] + t[i]);
    a[i] = av;
    k[i] = k[i] * (1.0f + (av - 1.0f) * k_a[c]);
  }
}

__global__ void k_vgate(const float* __restrict__ t, const float* __restrict__ v0,
                        const float* __restrict__ vf, float* __restrict__ v, int n) {
  int i = blockIdx.x * blockDim.x + threadIdx.x;
  if (i < n) {
    int c = i % CC;
    float s = sigmf(v0[c] + t[i]);
    v[i] = v[i] + (vf[i] - v[i]) * s;
  }
}

__global__ void k_kk(const float* __restrict__ k, const float* __restrict__ k_k,
                     float* __restrict__ kk, int n) {
  int g = blockIdx.x * blockDim.x + threadIdx.x;   // one thread per (row, head)
  if (g >= n) return;
  int row = g / HH, h = g % HH;
  size_t base = (size_t)row * CC + h * NH;
  float ss = 0.f;
  #pragma unroll 8
  for (int j = 0; j < NH; j++) { float t = k[base + j] * k_k[h * NH + j]; ss += t * t; }
  float inv = 1.0f / fmaxf(sqrtf(ss), 1e-12f);
  #pragma unroll 8
  for (int j = 0; j < NH; j++) kk[base + j] = k[base + j] * k_k[h * NH + j] * inv;
}

__global__ void k_add(const float* __restrict__ a, const float* __restrict__ b,
                      float* __restrict__ o, int n) {
  int i = blockIdx.x * blockDim.x + threadIdx.x;
  if (i < n) o[i] = a[i] + b[i];
}

__global__ void k_add_inplace(float* __restrict__ io, const float* __restrict__ b, int n) {
  int i = blockIdx.x * blockDim.x + threadIdx.x;
  if (i < n) io[i] += b[i];
}

__global__ void k_copy(float* __restrict__ d, const float* __restrict__ s, int n) {
  int i = blockIdx.x * blockDim.x + threadIdx.x;
  if (i < n) d[i] = s[i];
}

// ---------------- layer norm (row per block) ----------------
__global__ void __launch_bounds__(256) k_layernorm(const float* __restrict__ x,
                                                   const float* __restrict__ w,
                                                   const float* __restrict__ b,
                                                   float* __restrict__ y) {
  int row = blockIdx.x, tid = threadIdx.x;
  const float* xr = x + (size_t)row * CC;
  float s = 0.f, ss = 0.f;
  for (int c = tid; c < CC; c += 256) { float v = xr[c]; s += v; ss += v * v; }
  __shared__ float r1[256], r2[256];
  r1[tid] = s; r2[tid] = ss; __syncthreads();
  for (int st = 128; st > 0; st >>= 1) {
    if (tid < st) { r1[tid] += r1[tid + st]; r2[tid] += r2[tid + st]; }
    __syncthreads();
  }
  float mean = r1[0] / CC, var = r2[0] / CC - mean * mean;
  float inv = rsqrtf(var + 1e-5f);
  float* yr = y + (size_t)row * CC;
  for (int c = tid; c < CC; c += 256) yr[c] = (xr[c] - mean) * inv * w[c] + b[c];
}

// ---------------- time-shift mixes (6 bf16 outputs) ----------------
__global__ void k_prep_tmix(const float* __restrict__ xn,
    const float* __restrict__ mr, const float* __restrict__ mw, const float* __restrict__ mk,
    const float* __restrict__ mv, const float* __restrict__ ma, const float* __restrict__ mg,
    __bf16* __restrict__ xr, __bf16* __restrict__ xw, __bf16* __restrict__ xk,
    __bf16* __restrict__ xv, __bf16* __restrict__ xa, __bf16* __restrict__ xg, int n) {
  int i = blockIdx.x * blockDim.x + threadIdx.x;
  if (i >= n) return;
  int row = i / CC, c = i - row * CC, t = row % TT;
  float cur = xn[i];
  float xx = (t > 0 ? xn[i - CC] : 0.f) - cur;
  xr[i] = f2bf(cur + xx * mr[c]);
  xw[i] = f2bf(cur + xx * mw[c]);
  xk[i] = f2bf(cur + xx * mk[c]);
  xv[i] = f2bf(cur + xx * mv[c]);
  xa[i] = f2bf(cur + xx * ma[c]);
  xg[i] = f2bf(cur + xx * mg[c]);
}

__global__ void k_prep_cmix(const float* __restrict__ x2, const float* __restrict__ mix,
                            __bf16* __restrict__ out, int n) {
  int i = blockIdx.x * blockDim.x + threadIdx.x;
  if (i >= n) return;
  int row = i / CC, c = i - row * CC, t = row % TT;
  float cur = x2[i];
  float xx = (t > 0 ? x2[i - CC] : 0.f) - cur;
  out[i] = f2bf(cur + xx * mix[c]);
}

// ---------------- bf16 WMMA GEMM: C[M,N] = A[M,K] * B[K,N] ----------------
// block tile 128x128x32, 8 wave32 waves arranged 4(M) x 2(N); each wave 2x4 WMMA tiles.
__global__ void __launch_bounds__(256) k_gemm_bf16(
    const __bf16* __restrict__ A, const __bf16* __restrict__ B,
    float* __restrict__ Cm, int Mm, int Nn, int Kk) {
  __shared__ __bf16 As[128 * 40];   // [m][k], stride 40 keeps 16B alignment
  __shared__ __bf16 Bs[128 * 40];   // [n][k] (transposed stage)
  int tid = threadIdx.x;
  int lane = tid & 31, wave = tid >> 5;
  int wm = wave & 3;                // 32-row slab
  int wn = wave >> 2;               // 64-col slab
  int lm = lane & 15;
  int half8  = (lane >= 16) ? 8  : 0;
  int half16 = (lane >= 16) ? 16 : 0;
  int bm = blockIdx.y * 128;
  int bn = blockIdx.x * 128;

  v8f acc[2][4] = {};

  for (int k0 = 0; k0 < Kk; k0 += 32) {
    __syncthreads();
    // stage A: 128x32
    #pragma unroll
    for (int l = 0; l < 2; l++) {
      int cidx = tid + l * 256;
      int arow = cidx >> 2;
      int acol = (cidx & 3) * 8;
      v8bf val = *(const v8bf*)(A + (size_t)(bm + arow) * Kk + k0 + acol);
      *(v8bf*)&As[arow * 40 + acol] = val;
    }
    // stage B transposed: Bs[n][k]
    #pragma unroll
    for (int l = 0; l < 2; l++) {
      int cidx = tid + l * 256;
      int krow = cidx >> 4;
      int ncol = (cidx & 15) * 8;
      int ng = bn + ncol;
      if (ng + 8 <= Nn) {
        v8bf val = *(const v8bf*)(B + (size_t)(k0 + krow) * Nn + ng);
        #pragma unroll
        for (int e = 0; e < 8; e++) Bs[(ncol + e) * 40 + krow] = val[e];
      } else {
        #pragma unroll
        for (int e = 0; e < 8; e++) {
          __bf16 v = (ng + e < Nn) ? B[(size_t)(k0 + krow) * Nn + ng + e] : f2bf(0.f);
          Bs[(ncol + e) * 40 + krow] = v;
        }
      }
    }
    if (k0 + 32 < Kk)
      __builtin_prefetch(A + (size_t)(bm + (tid >> 1)) * Kk + k0 + 32, 0, 0);
    __syncthreads();

    // fragments: A-frag = two 8-elem K runs; B-frag = two 8-elem K runs (same order)
    v16bf af[2];
    #pragma unroll
    for (int mi = 0; mi < 2; mi++) {
      int mrow = wm * 32 + mi * 16 + lm;
      v8bf lo = *(const v8bf*)&As[mrow * 40 + half8];
      v8bf hi = *(const v8bf*)&As[mrow * 40 + 16 + half8];
      #pragma unroll
      for (int e = 0; e < 8; e++) { af[mi][e] = lo[e]; af[mi][e + 8] = hi[e]; }
    }
    v16bf bq[4];
    #pragma unroll
    for (int ni = 0; ni < 4; ni++) {
      int nrow = wn * 64 + ni * 16 + lm;
      v8bf lo = *(const v8bf*)&Bs[nrow * 40 + half8];
      v8bf hi = *(const v8bf*)&Bs[nrow * 40 + 16 + half8];
      #pragma unroll
      for (int e = 0; e < 8; e++) { bq[ni][e] = lo[e]; bq[ni][e + 8] = hi[e]; }
    }
    #pragma unroll
    for (int mi = 0; mi < 2; mi++)
      #pragma unroll
      for (int ni = 0; ni < 4; ni++)
        acc[mi][ni] = __builtin_amdgcn_wmma_f32_16x16x32_bf16(
            false, af[mi], false, bq[ni], (short)0, acc[mi][ni], false, false);
  }

  #pragma unroll
  for (int mi = 0; mi < 2; mi++) {
    #pragma unroll
    for (int ni = 0; ni < 4; ni++) {
      int ng = bn + wn * 64 + ni * 16 + lm;
      if (ng < Nn) {
        #pragma unroll
        for (int i = 0; i < 8; i++) {
          int mg = bm + wm * 32 + mi * 16 + i + ((lane >= 16) ? 8 : 0);
          Cm[(size_t)mg * Nn + ng] = acc[mi][ni][i];
        }
      }
    }
  }
}

// ---------------- RWKV-7 sequential scan ----------------
// one block per (b,h); 64 threads; thread i owns state row S[i][0..63] in VGPRs.
__global__ void __launch_bounds__(64) k_rwkv7_scan(
    const float* __restrict__ r, const float* __restrict__ wlog,
    const float* __restrict__ k, const float* __restrict__ v,
    const float* __restrict__ kk, const float* __restrict__ a,
    float* __restrict__ y) {
  int bh = blockIdx.x;
  int b = bh >> 5, h = bh & 31;
  int i = threadIdx.x;
  float s[NH];
  #pragma unroll
  for (int j = 0; j < NH; j++) s[j] = 0.f;
  __shared__ float lr[NH], lk[NH], lv[NH], la[NH], lb[NH], ld[NH];
  for (int t = 0; t < TT; t++) {
    size_t idx = ((size_t)(b * TT + t)) * CC + h * NH + i;
    float kkv = kk[idx];
    la[i] = -kkv;
    lb[i] = kkv * a[idx];
    lr[i] = r[idx];
    lk[i] = k[idx];
    lv[i] = v[idx];
    ld[i] = __expf(-__expf(wlog[idx]));
    __syncthreads();
    float sa = 0.f;
    #pragma unroll 8
    for (int j = 0; j < NH; j++) sa += s[j] * la[j];
    float vi = lv[i];
    float yv = 0.f;
    #pragma unroll 8
    for (int j = 0; j < NH; j++) {
      float sj = s[j] * ld[j] + sa * lb[j] + vi * lk[j];
      s[j] = sj;
      yv += sj * lr[j];
    }
    y[idx] = yv;
    __syncthreads();
  }
}

// ---------------- group norm + r.k residual + gate, bf16 out ----------------
__global__ void __launch_bounds__(256) k_gn_out(
    const float* __restrict__ y, const float* __restrict__ r, const float* __restrict__ k,
    const float* __restrict__ v, const float* __restrict__ g,
    const float* __restrict__ gn_w, const float* __restrict__ gn_b,
    const float* __restrict__ r_k, __bf16* __restrict__ yg) {
  int row = blockIdx.x, tid = threadIdx.x;
  size_t base = (size_t)row * CC;
  int c0 = tid * 8;
  float yv[8];
  float ps = 0.f, pss = 0.f, prk = 0.f;
  #pragma unroll
  for (int e = 0; e < 8; e++) {
    int c = c0 + e;
    float t = y[base + c]; yv[e] = t; ps += t; pss += t * t;
    prk += r[base + c] * k[base + c] * r_k[c];
  }
  __shared__ float s1[256], s2[256], s3[256];
  __shared__ float hm[HH], hi[HH], hr[HH];
  s1[tid] = ps; s2[tid] = pss; s3[tid] = prk;
  __syncthreads();
  if (tid < HH) {
    float a = 0.f, b2 = 0.f, cc = 0.f;
    #pragma unroll
    for (int e = 0; e < 8; e++) { a += s1[tid * 8 + e]; b2 += s2[tid * 8 + e]; cc += s3[tid * 8 + e]; }
    float mean = a / NH, var = b2 / NH - mean * mean;
    hm[tid] = mean;
    hi[tid] = rsqrtf(var + 6.4e-4f);   // GN_EPS = 1e-5 * 8^2
    hr[tid] = cc;
  }
  __syncthreads();
  int h = tid >> 3;
  #pragma unroll
  for (int e = 0; e < 8; e++) {
    int c = c0 + e;
    float t = (yv[e] - hm[h]) * hi[h] * gn_w[c] + gn_b[c];
    t += hr[h] * v[base + c];
    yg[base + c] = f2bf(t * g[base + c]);
  }
}

// ---------------- host launch ----------------
extern "C" void kernel_launch(void* const* d_in, const int* in_sizes, int n_in,
                              void* d_out, int out_size, void* d_ws, size_t ws_size,
                              hipStream_t stream) {
  const float* x      = (const float*)d_in[0];
  const float* vfirst = (const float*)d_in[1];
  const float* ln1_w  = (const float*)d_in[2];
  const float* ln1_b  = (const float*)d_in[3];
  const float* ln2_w  = (const float*)d_in[4];
  const float* ln2_b  = (const float*)d_in[5];
  const float* x_r    = (const float*)d_in[6];
  const float* x_w    = (const float*)d_in[7];
  const float* x_k    = (const float*)d_in[8];
  const float* x_v    = (const float*)d_in[9];
  const float* x_a    = (const float*)d_in[10];
  const float* x_g    = (const float*)d_in[11];
  const float* w0     = (const float*)d_in[12];
  const float* w1     = (const float*)d_in[13];
  const float* w2     = (const float*)d_in[14];
  const float* a0     = (const float*)d_in[15];
  const float* a1     = (const float*)d_in[16];
  const float* a2     = (const float*)d_in[17];
  const float* v0     = (const float*)d_in[18];
  const float* v1     = (const float*)d_in[19];
  const float* v2     = (const float*)d_in[20];
  const float* g1     = (const float*)d_in[21];
  const float* g2     = (const float*)d_in[22];
  const float* k_kw   = (const float*)d_in[23];
  const float* k_aw   = (const float*)d_in[24];
  const float* r_kw   = (const float*)d_in[25];
  const float* W_r    = (const float*)d_in[26];
  const float* W_k    = (const float*)d_in[27];
  const float* W_v    = (const float*)d_in[28];
  const float* W_o    = (const float*)d_in[29];
  const float* gn_w   = (const float*)d_in[30];
  const float* gn_b   = (const float*)d_in[31];
  const float* ffn_xk = (const float*)d_in[32];
  const float* W_key  = (const float*)d_in[33];
  const float* W_val  = (const float*)d_in[34];

  const size_t MC  = (size_t)MM * CC;        // 6291456
  const size_t MC4 = (size_t)MM * C4;        // 25165824
  char* ws = (char*)d_ws;
  size_t off = 0;
  auto alloc = [&](size_t bytes) -> char* {
    char* p = ws + off;
    off += (bytes + 255) & ~(size_t)255;
    return p;
  };

  // bf16 weights
  __bf16* wr_bf   = (__bf16*)alloc((size_t)CC * CC * 2);
  __bf16* wk_bf   = (__bf16*)alloc((size_t)CC * CC * 2);
  __bf16* wv_bf   = (__bf16*)alloc((size_t)CC * CC * 2);
  __bf16* wo_bf   = (__bf16*)alloc((size_t)CC * CC * 2);
  __bf16* wkey_bf = (__bf16*)alloc((size_t)CC * C4 * 2);
  __bf16* wval_bf = (__bf16*)alloc((size_t)C4 * CC * 2);
  __bf16* w1_bf   = (__bf16*)alloc((size_t)CC * 96 * 2);
  __bf16* w2_bf   = (__bf16*)alloc((size_t)96 * CC * 2);
  __bf16* a1_bf   = (__bf16*)alloc((size_t)CC * 96 * 2);
  __bf16* a2_bf   = (__bf16*)alloc((size_t)96 * CC * 2);
  __bf16* v1_bf   = (__bf16*)alloc((size_t)CC * 64 * 2);
  __bf16* v2_bf   = (__bf16*)alloc((size_t)64 * CC * 2);
  __bf16* g1_bf   = (__bf16*)alloc((size_t)CC * 256 * 2);
  __bf16* g2_bf   = (__bf16*)alloc((size_t)256 * CC * 2);
  // activations
  float*  xn      = (float*)alloc(MC * 4);
  __bf16* xr_bf   = (__bf16*)alloc(MC * 2);
  __bf16* xw_bf   = (__bf16*)alloc(MC * 2);
  __bf16* xk_bf   = (__bf16*)alloc(MC * 2);
  __bf16* xv_bf   = (__bf16*)alloc(MC * 2);
  __bf16* xa_bf   = (__bf16*)alloc(MC * 2);
  __bf16* xg_bf   = (__bf16*)alloc(MC * 2);
  float*  rbuf    = (float*)alloc(MC * 4);
  float*  kbuf    = (float*)alloc(MC * 4);
  float*  vbuf    = (float*)alloc(MC * 4);
  float*  wlog    = (float*)alloc(MC * 4);
  float*  abuf    = (float*)alloc(MC * 4);
  float*  gbuf    = (float*)alloc(MC * 4);
  float*  kkbuf   = (float*)alloc(MC * 4);
  float*  ybuf    = (float*)alloc(MC * 4);
  float*  tmpC    = (float*)alloc(MC * 4);
  float*  tmp96   = (float*)alloc((size_t)MM * 96 * 4);
  __bf16* tmp96b  = (__bf16*)alloc((size_t)MM * 96 * 2);
  float*  tmp64   = (float*)alloc((size_t)MM * 64 * 4);
  __bf16* tmp64b  = (__bf16*)alloc((size_t)MM * 64 * 2);
  float*  tmpG    = (float*)alloc((size_t)MM * 256 * 4);
  __bf16* tmpGb   = (__bf16*)alloc((size_t)MM * 256 * 2);
  __bf16* yg_bf   = (__bf16*)alloc(MC * 2);
  __bf16* kmix_bf = (__bf16*)alloc(MC * 2);
  float*  kf_lin  = (float*)alloc(MC4 * 4);
  __bf16* kf_bf   = (__bf16*)alloc(MC4 * 2);

  float* xout = (float*)d_out;            // first MC: x; next MC: v_first

  auto ew = [&](int n) { return dim3((n + 255) / 256); };
  auto gemm = [&](const __bf16* A, const __bf16* B, float* C, int N, int K) {
    dim3 grid((N + 127) / 128, MM / 128);
    k_gemm_bf16<<<grid, 256, 0, stream>>>(A, B, C, MM, N, K);
  };

  // weight conversion f32 -> bf16
  k_f32_to_bf16<<<ew(CC * CC), 256, 0, stream>>>(W_r,   wr_bf,   CC * CC);
  k_f32_to_bf16<<<ew(CC * CC), 256, 0, stream>>>(W_k,   wk_bf,   CC * CC);
  k_f32_to_bf16<<<ew(CC * CC), 256, 0, stream>>>(W_v,   wv_bf,   CC * CC);
  k_f32_to_bf16<<<ew(CC * CC), 256, 0, stream>>>(W_o,   wo_bf,   CC * CC);
  k_f32_to_bf16<<<ew(CC * C4), 256, 0, stream>>>(W_key, wkey_bf, CC * C4);
  k_f32_to_bf16<<<ew(CC * C4), 256, 0, stream>>>(W_val, wval_bf, CC * C4);
  k_f32_to_bf16<<<ew(CC * 96), 256, 0, stream>>>(w1, w1_bf, CC * 96);
  k_f32_to_bf16<<<ew(CC * 96), 256, 0, stream>>>(w2, w2_bf, CC * 96);
  k_f32_to_bf16<<<ew(CC * 96), 256, 0, stream>>>(a1, a1_bf, CC * 96);
  k_f32_to_bf16<<<ew(CC * 96), 256, 0, stream>>>(a2, a2_bf, CC * 96);
  k_f32_to_bf16<<<ew(CC * 64), 256, 0, stream>>>(v1, v1_bf, CC * 64);
  k_f32_to_bf16<<<ew(CC * 64), 256, 0, stream>>>(v2, v2_bf, CC * 64);
  k_f32_to_bf16<<<ew(CC * 256), 256, 0, stream>>>(g1, g1_bf, CC * 256);
  k_f32_to_bf16<<<ew(CC * 256), 256, 0, stream>>>(g2, g2_bf, CC * 256);

  // ---- time mix ----
  k_layernorm<<<MM, 256, 0, stream>>>(x, ln1_w, ln1_b, xn);
  k_prep_tmix<<<ew((int)MC), 256, 0, stream>>>(xn, x_r, x_w, x_k, x_v, x_a, x_g,
      xr_bf, xw_bf, xk_bf, xv_bf, xa_bf, xg_bf, (int)MC);

  gemm(xr_bf, wr_bf, rbuf, CC, CC);       // r
  gemm(xk_bf, wk_bf, kbuf, CC, CC);       // k (pre-update)
  gemm(xv_bf, wv_bf, vbuf, CC, CC);       // v (pre-gate)

  // w path: wlog = -softplus(-(w0 + tanh(xw@w1)@w2)) - 0.5
  gemm(xw_bf, w1_bf, tmp96, 96, CC);
  k_tanh_bf<<<ew(MM * 96), 256, 0, stream>>>(tmp96, tmp96b, MM * 96);
  gemm(tmp96b, w2_bf, tmpC, CC, 96);
  k_wlog<<<ew((int)MC), 256, 0, stream>>>(tmpC, w0, wlog, (int)MC);

  // a path: a = sigmoid(a0 + (xa@a1)@a2); then kk (from raw k) then k update
  gemm(xa_bf, a1_bf, tmp96, 96, CC);
  k_f32_to_bf16<<<ew(MM * 96), 256, 0, stream>>>(tmp96, tmp96b, MM * 96);
  gemm(tmp96b, a2_bf, tmpC, CC, 96);
  k_kk<<<ew(MM * HH), 256, 0, stream>>>(kbuf, k_kw, kkbuf, MM * HH);
  k_a_and_k<<<ew((int)MC), 256, 0, stream>>>(tmpC, a0, k_aw, abuf, kbuf, (int)MC);

  // v gate: v += (v_first - v)*sigmoid(v0 + (xv@v1)@v2)
  gemm(xv_bf, v1_bf, tmp64, 64, CC);
  k_f32_to_bf16<<<ew(MM * 64), 256, 0, stream>>>(tmp64, tmp64b, MM * 64);
  gemm(tmp64b, v2_bf, tmpC, CC, 64);
  k_vgate<<<ew((int)MC), 256, 0, stream>>>(tmpC, v0, vfirst, vbuf, (int)MC);

  // g path: g = sigmoid(xg@g1)@g2
  gemm(xg_bf, g1_bf, tmpG, 256, CC);
  k_sigmoid_bf<<<ew(MM * 256), 256, 0, stream>>>(tmpG, tmpGb, MM * 256);
  gemm(tmpGb, g2_bf, gbuf, CC, 256);

  // sequential RWKV-7 scan
  k_rwkv7_scan<<<TB * HH, 64, 0, stream>>>(rbuf, wlog, kbuf, vbuf, kkbuf, abuf, ybuf);

  // group-norm + r.k residual + gate
  k_gn_out<<<MM, 256, 0, stream>>>(ybuf, rbuf, kbuf, vbuf, gbuf, gn_w, gn_b, r_kw, yg_bf);

  // output projection + residual
  gemm(yg_bf, wo_bf, tmpC, CC, CC);
  k_add<<<ew((int)MC), 256, 0, stream>>>(x, tmpC, xout, (int)MC);

  // ---- channel mix ----
  k_layernorm<<<MM, 256, 0, stream>>>(xout, ln2_w, ln2_b, xn);
  k_prep_cmix<<<ew((int)MC), 256, 0, stream>>>(xn, ffn_xk, kmix_bf, (int)MC);
  gemm(kmix_bf, wkey_bf, kf_lin, C4, CC);
  k_relusq_bf<<<ew((int)MC4), 256, 0, stream>>>(kf_lin, kf_bf, (int)MC4);
  gemm(kf_bf, wval_bf, tmpC, CC, C4);
  k_add_inplace<<<ew((int)MC), 256, 0, stream>>>(xout, tmpC, (int)MC);

  // second output: v_first passthrough
  k_copy<<<ew((int)MC), 256, 0, stream>>>(xout + MC, vfirst, (int)MC);
}
